// ControllerDecoder_18245021073459
// MI455X (gfx1250) — matrix operand order repaired and checked by reading, not compiled, not run on hardware
//
#include <hip/hip_runtime.h>

#define IN_SIZE  65536
#define OUT_SIZE 65536
#define LATENT   32
#define HIDDEN   128

// output layout: [w1 (65536*8)] [w2 (64)] [w3 (8*65536)]
#define W2_OUT_OFF (IN_SIZE * 8)
#define W3_OUT_OFF (IN_SIZE * 8 + 64)

typedef _Float16 v16h __attribute__((ext_vector_type(16)));
typedef float    v8f  __attribute__((ext_vector_type(8)));

// -------------------------------------------------------------------------
// Kernel 1: setup.
//  - pre8 = latent[:8] @ W1[:32]  (8x128)
//  - emits w2 (8x8) directly
//  - stores t3 (layer-3 bases) to ws in [k][i] layout (for w3 b128 reads)
//  - pre-packs W1[:32] as f16 B-fragments in per-lane WMMA layout:
//      bPack[nt*512 + lane*16 + t] = (f16) W1[K(t, lane>>4)*128 + nt*16 + (lane&15)], K = t<8 ? t+8g : 16+(t-8)+8g
// -------------------------------------------------------------------------
__global__ __launch_bounds__(128) void hyper_setup(
    const float* __restrict__ latent, const float* __restrict__ W1,
    const float* __restrict__ b1, const float* __restrict__ W2,
    const float* __restrict__ b2, float* __restrict__ out,
    float* __restrict__ t3_ws, _Float16* __restrict__ bPack)
{
    __shared__ float t2s[8 * HIDDEN];
    __shared__ float us[HIDDEN];
    __shared__ float w2s[HIDDEN];

    const int k = threadIdx.x;  // 0..127 == hidden index
    const float v  = W1[32 * HIDDEN + k];  // pi coefficient
    const float u  = W1[33 * HIDDEN + k];  // pj coefficient
    const float s  = W1[34 * HIDDEN + k];  // layer_id coefficient
    const float bb = b1[k];
    us[k]  = u;
    w2s[k] = W2[k];

    for (int i = 0; i < 8; ++i) {
        float pre = 0.0f;
        #pragma unroll
        for (int l = 0; l < LATENT; ++l)
            pre = fmaf(latent[i * LATENT + l], W1[l * HIDDEN + k], pre);
        const float pi = (float)i * 0.125f;         // arange(8)/8
        const float common = fmaf(pi, v, pre) + bb;
        t2s[i * HIDDEN + k] = fmaf(2.0f, s, common);   // layer_id = 2
        t3_ws[k * 8 + i]    = fmaf(3.0f, s, common);   // layer_id = 3, [k][i]
    }

    // pre-pack B fragments (4096 halves), loop-invariant for hyper_w1
    for (int d = k; d < 8 * 512; d += 128) {
        const int nt   = d >> 9;
        const int r    = d & 511;
        const int lane = r >> 4;
        const int t    = r & 15;
        const int mm   = lane & 15;
        const int gg   = lane >> 4;
        bPack[d] = (_Float16)W1[((t < 8) ? (t + 8 * gg) : (16 + (t - 8) + 8 * gg)) * HIDDEN + nt * 16 + mm];
    }

    __syncthreads();

    // 64 outputs of w2
    if (k < 64) {
        const int i = k >> 3, j = k & 7;
        const float pj = (float)j * 0.125f;
        float acc = b2[0];
        #pragma unroll 4
        for (int kk = 0; kk < HIDDEN; ++kk) {
            const float z = fmaf(pj, us[kk], t2s[i * HIDDEN + kk]);
            acc = fmaf(fmaxf(z, 0.0f), w2s[kk], acc);
        }
        out[W2_OUT_OFF + i * 8 + j] = acc;
    }
}

// -------------------------------------------------------------------------
// Kernel 2: w1. Per wave: 16 rows. pre(16x128) = latentTile(16x32) @ W1[:32]
// via 8x v_wmma_f32_16x16x32_f16; B fragments are pre-packed f16 (b128
// loads, no converts). C tiles go to bank-padded LDS; each lane then
// computes 4 (row, j) relu-dot outputs with a b128 coefficient read per k.
// -------------------------------------------------------------------------
__global__ __launch_bounds__(128) void hyper_w1(
    const float* __restrict__ latent, const float* __restrict__ W1,
    const float* __restrict__ b1, const float* __restrict__ W2,
    const float* __restrict__ b2, const _Float16* __restrict__ bPack,
    float* __restrict__ out)
{
    __shared__ float4 cpk[HIDDEN];      // {v, u, c0, w2} per hidden k
    __shared__ float  aS[4][16 * 33];   // latent tile per wave, pad 32->33
    __shared__ float  preT[4][16 * 132];// pre tile per wave, pad 128->132

    const int tid  = threadIdx.x;
    const int wave = tid >> 5;
    const int lane = tid & 31;
    const int m    = lane & 15;   // row-in-tile (A/C) or column (B)
    const int g    = lane >> 4;   // lane half-group

    // coefficient staging (blockDim == 128 == HIDDEN)
    cpk[tid] = make_float4(W1[32 * HIDDEN + tid],            // v (pi coef)
                           W1[33 * HIDDEN + tid],            // u (pj coef)
                           W1[34 * HIDDEN + tid] + b1[tid],  // c0 = 1*s + b1
                           W2[tid]);

    const int row0 = blockIdx.x * 64 + wave * 16;

    // stage 16x32 latent tile (coalesced: 512 contiguous floats)
    const float* lsrc = latent + (size_t)row0 * LATENT;
    #pragma unroll
    for (int it = 0; it < 16; ++it)
        aS[wave][it * 33 + lane] = lsrc[it * 32 + lane];

    __syncthreads();

    // A fragment: 16-bit A 16x32 layout (VGPR0..3: K = h + 8g,
    // VGPR4..7: K = 16 + (h-8) + 8g), M = lane&15.
    v16h afrag;
    #pragma unroll
    for (int h = 0; h < 16; ++h) {
        const int K = (h < 8) ? (h + 8 * g) : (16 + (h - 8) + 8 * g);
        afrag[h] = (_Float16)aS[wave][m * 33 + K];
    }

    // speculative prefetch of next block's latent tile
    if (blockIdx.x + 1 < gridDim.x)
        __builtin_prefetch(latent + (size_t)(row0 + 64) * LATENT, 0, 3);

    // 8 column tiles of pre = tile @ W1[:32]
    #pragma unroll
    for (int nt = 0; nt < 8; ++nt) {
        const int n0 = nt * 16;
        // pre-packed B fragment: one 32B vector load per lane
        const v16h bfrag = *(const v16h*)(bPack + (nt << 9) + (lane << 4));

        v8f c = {};
        c = __builtin_amdgcn_wmma_f32_16x16x32_f16(
                /*neg_a=*/false, afrag, /*neg_b=*/false, bfrag,
                /*c_mod=*/(short)0, c, /*reuse_a=*/false, /*reuse_b=*/false);

        // C layout: VGPR r -> M = r + 8g, N = lane&15
        #pragma unroll
        for (int r = 0; r < 8; ++r)
            preT[wave][(r + 8 * g) * 132 + n0 + m] = c[r];
    }

    __syncthreads();

    // phase 2: lane handles row m, j in [4g, 4g+4)
    const int   jbase = g * 4;
    const int   row   = row0 + m;
    const float pi    = (float)row * (1.0f / (float)IN_SIZE);
    const float pj0   = (float)(jbase + 0) * 0.125f;
    const float pj1   = (float)(jbase + 1) * 0.125f;
    const float pj2   = (float)(jbase + 2) * 0.125f;
    const float pj3   = (float)(jbase + 3) * 0.125f;
    const float b2v   = b2[0];
    float acc0 = b2v, acc1 = b2v, acc2 = b2v, acc3 = b2v;

    #pragma unroll 4
    for (int k = 0; k < HIDDEN; ++k) {
        const float4 cf  = cpk[k];                       // ds_load_b128
        const float base = fmaf(pi, cf.x, preT[wave][m * 132 + k] + cf.z);
        acc0 = fmaf(fmaxf(fmaf(pj0, cf.y, base), 0.0f), cf.w, acc0);
        acc1 = fmaf(fmaxf(fmaf(pj1, cf.y, base), 0.0f), cf.w, acc1);
        acc2 = fmaf(fmaxf(fmaf(pj2, cf.y, base), 0.0f), cf.w, acc2);
        acc3 = fmaf(fmaxf(fmaf(pj3, cf.y, base), 0.0f), cf.w, acc3);
    }

    float4 res = make_float4(acc0, acc1, acc2, acc3);
    *(float4*)(out + (size_t)row * 8 + jbase) = res;  // 16B aligned
}

// -------------------------------------------------------------------------
// Kernel 3: w3. One thread per output column j; 8 row accumulators.
// t3 table in [k][i] layout -> two ds_load_b128 per k; stores coalesced.
// -------------------------------------------------------------------------
__global__ __launch_bounds__(256) void hyper_w3(
    const float* __restrict__ W1, const float* __restrict__ W2,
    const float* __restrict__ b2, const float* __restrict__ t3_ws,
    float* __restrict__ out)
{
    __shared__ float4 t3s4[HIDDEN * 2]; // t3 in [k][i] layout, float4 pairs
    __shared__ float2 upk[HIDDEN];      // {u, w2} per hidden k

    const int tid = threadIdx.x;
    if (tid < HIDDEN * 2)
        t3s4[tid] = ((const float4*)t3_ws)[tid];
    if (tid < HIDDEN)
        upk[tid] = make_float2(W1[33 * HIDDEN + tid], W2[tid]);
    __syncthreads();

    const int   j   = blockIdx.x * 256 + tid;
    const float pj  = (float)j * (1.0f / (float)OUT_SIZE);
    const float b2v = b2[0];

    float acc[8];
    #pragma unroll
    for (int i = 0; i < 8; ++i) acc[i] = b2v;

    #pragma unroll 2
    for (int k = 0; k < HIDDEN; ++k) {
        const float2 c  = upk[k];                 // ds_load_b64 (broadcast)
        const float  p  = pj * c.x;
        const float  wk = c.y;
        const float4 lo = t3s4[k * 2 + 0];        // ds_load_b128 (broadcast)
        const float4 hi = t3s4[k * 2 + 1];
        acc[0] = fmaf(fmaxf(lo.x + p, 0.0f), wk, acc[0]);
        acc[1] = fmaf(fmaxf(lo.y + p, 0.0f), wk, acc[1]);
        acc[2] = fmaf(fmaxf(lo.z + p, 0.0f), wk, acc[2]);
        acc[3] = fmaf(fmaxf(lo.w + p, 0.0f), wk, acc[3]);
        acc[4] = fmaf(fmaxf(hi.x + p, 0.0f), wk, acc[4]);
        acc[5] = fmaf(fmaxf(hi.y + p, 0.0f), wk, acc[5]);
        acc[6] = fmaf(fmaxf(hi.z + p, 0.0f), wk, acc[6]);
        acc[7] = fmaf(fmaxf(hi.w + p, 0.0f), wk, acc[7]);
    }

    #pragma unroll
    for (int i = 0; i < 8; ++i)
        out[(size_t)W3_OUT_OFF + (size_t)i * OUT_SIZE + j] = acc[i];
}

// -------------------------------------------------------------------------
extern "C" void kernel_launch(void* const* d_in, const int* in_sizes, int n_in,
                              void* d_out, int out_size, void* d_ws, size_t ws_size,
                              hipStream_t stream) {
    (void)in_sizes; (void)n_in; (void)out_size; (void)ws_size;
    const float* latent = (const float*)d_in[0];
    const float* W1     = (const float*)d_in[1];
    const float* b1     = (const float*)d_in[2];
    const float* W2     = (const float*)d_in[3];
    const float* b2     = (const float*)d_in[4];
    float* out = (float*)d_out;

    // ws layout: [t3: 1024 floats (4KB)] [bPack: 4096 halves (8KB)]
    float*    t3_ws = (float*)d_ws;
    _Float16* bPack = (_Float16*)((char*)d_ws + 4096);

    hyper_setup<<<1, 128, 0, stream>>>(latent, W1, b1, W2, b2, out, t3_ws, bPack);
    hyper_w1<<<IN_SIZE / 64, 128, 0, stream>>>(latent, W1, b1, W2, b2, bPack, out);
    hyper_w3<<<OUT_SIZE / 256, 256, 0, stream>>>(W1, W2, b2, t3_ws, out);
}
